// GCN2_5488968204991
// MI455X (gfx1250) — matrix operand order repaired and checked by reading, not compiled, not run on hardware
//
#include <hip/hip_runtime.h>
#include <hip/hip_bf16.h>

typedef __attribute__((ext_vector_type(16))) _Float16 v16h;
typedef __attribute__((ext_vector_type(8)))  _Float16 v8h;
typedef __attribute__((ext_vector_type(4)))  _Float16 v4h;
typedef __attribute__((ext_vector_type(8)))  float    v8f;
typedef __attribute__((ext_vector_type(4)))  float    v4f;

#define NODES 50000
#define EDGES 800000
#define HID   128
#define FIN   11
#define NCLS  19
#define NGRAPH 2048

__device__ __forceinline__ void atomAddF(float* p, float v) {
    unsafeAtomicAdd(p, v);   // -> global_atomic_add_f32 (hardware fp32 atomic in L2)
}

// ---------------- degree / dinv ----------------
__global__ void k_deg_init(float* deg) {
    int t = blockIdx.x * blockDim.x + threadIdx.x;
    if (t < NODES) deg[t] = 1.0f;                 // self-loop included
}
__global__ void k_deg_count(const int* __restrict__ dst, float* deg) {
    int t = blockIdx.x * blockDim.x + threadIdx.x;
    if (t < EDGES) atomAddF(&deg[dst[t]], 1.0f);
}
__global__ void k_dinv(const float* __restrict__ deg, float* dinv) {
    int t = blockIdx.x * blockDim.x + threadIdx.x;
    if (t < NODES) dinv[t] = rsqrtf(deg[t]);
}

// ---------------- layer-1 dense (K=11, scalar f32) ----------------
__global__ void k_gemm_l1(const float* __restrict__ x, const float* __restrict__ W1,
                          float* __restrict__ H) {
    int t = blockIdx.x * blockDim.x + threadIdx.x;
    if (t >= NODES * HID) return;
    int node = t >> 7, j = t & 127;
    const float* xr = x + node * FIN;
    float s = 0.f;
#pragma unroll
    for (int k = 0; k < FIN; ++k) s = fmaf(xr[k], W1[k * HID + j], s);
    H[t] = s;
}

// ---------------- pack weight [128x128] f32 -> WMMA B-fragment order f16 ----
// Bp index t = (((kb*8 + ntile)*32 + lane)*16 + j)
// lane<16: col = ntile*16+lane, K = kb*32 + {0..7,16..23}[j]      (j = 0..15)
// lane>=16: col = ntile*16+lane-16, K = kb*32 + {8..15,24..31}[j]
__global__ void k_pack_b(const float* __restrict__ W, _Float16* __restrict__ Bp) {
    int t = blockIdx.x * blockDim.x + threadIdx.x;
    if (t >= HID * HID) return;
    int j    = t & 15;
    int lane = (t >> 4) & 31;
    int nt   = (t >> 9) & 7;
    int kb   = t >> 12;
    int k = kb * 32 + (lane >> 4) * 8 + (j >> 3) * 16 + (j & 7);
    int n = nt * 16 + (lane & 15);
    Bp[t] = (_Float16)W[k * HID + n];
}

// ---------------- WMMA GEMM: C[M x 128] = A[M x 128]f16 * Bp(LDS) ---------
// one wave -> 16 rows x 128 cols; 8 waves / block; B staged in LDS once
__global__ __launch_bounds__(256) void k_gemm_wmma(const _Float16* __restrict__ A,
                                                   const _Float16* __restrict__ Bp,
                                                   float* __restrict__ C, int mtiles) {
    __shared__ _Float16 Bs[HID * HID];            // 32 KB of 320 KB WGP LDS
    // cooperative copy: 2048 x 16B chunks, 8 per thread
#pragma unroll
    for (int i = threadIdx.x; i < (HID * HID) / 8; i += 256)
        ((v8h*)Bs)[i] = ((const v8h*)Bp)[i];
    __syncthreads();

    int wave  = threadIdx.x >> 5;
    int lane  = threadIdx.x & 31;
    int mtile = blockIdx.x * 8 + wave;
    if (mtile >= mtiles) return;                  // wave-uniform: EXEC stays all-1s
    int hi  = lane >> 4;
    int row = mtile * 16 + (lane & 15);

    v8f acc[8] = {};
    const _Float16* bl = Bs + lane * 16;          // per-lane LDS base
#pragma unroll
    for (int kb = 0; kb < 4; ++kb) {
        // A fragment per ISA layout: lanes0-15 K={0..7,16..23}, lanes16-31 K={8..15,24..31}
        const _Float16* arow = A + (size_t)row * HID + kb * 32 + hi * 8;
        union { v16h v; v8h h[2]; } a;
        a.h[0] = *(const v8h*)(arow);
        a.h[1] = *(const v8h*)(arow + 16);
#pragma unroll
        for (int n = 0; n < 8; ++n) {
            v16h b = *(const v16h*)(bl + (kb * 8 + n) * 32 * 16);   // ds_load
            acc[n] = __builtin_amdgcn_wmma_f32_16x16x32_f16(
                false, a.v, false, b, (short)0, acc[n], false, false);
        }
    }
    // C/D layout: VGPR r -> row r (lanes0-15) / r+8 (lanes16-31), col = lane&15
    float* crow = C + (size_t)(mtile * 16 + hi * 8) * HID + (lane & 15);
#pragma unroll
    for (int n = 0; n < 8; ++n)
#pragma unroll
        for (int r = 0; r < 8; ++r)
            crow[(size_t)r * HID + n * 16] = acc[n][r];
}

// ---------------- self-loop term + bias:  Aout = H*dinv^2 + b (float4) ----
__global__ void k_self_bias(const float* __restrict__ H, const float* __restrict__ dinv,
                            const float* __restrict__ b, float* __restrict__ Aout) {
    int t = blockIdx.x * blockDim.x + threadIdx.x;
    if (t >= NODES * (HID / 4)) return;
    int node = t >> 5, j4 = t & 31;
    float di = dinv[node];
    float dd = di * di;
    v4f h  = ((const v4f*)H)[t];
    v4f bb = ((const v4f*)b)[j4];
    v4f r;
#pragma unroll
    for (int i = 0; i < 4; ++i) r[i] = fmaf(h[i], dd, bb[i]);
    ((v4f*)Aout)[t] = r;
}

// ---------------- edge scatter: Aout[dst] += H[src] * dinv[s]*dinv[d] -----
// one wave per edge; lane gathers contiguous float4, 4 fp32 atomics
__global__ __launch_bounds__(256) void k_edge_agg(const int* __restrict__ src,
                                                  const int* __restrict__ dst,
                                                  const float* __restrict__ dinv,
                                                  const float* __restrict__ H,
                                                  float* __restrict__ Aout) {
    int t = blockIdx.x * blockDim.x + threadIdx.x;
    int e = t >> 5;
    if (e >= EDGES) return;
    int lane = t & 31;
    int s = src[e], d = dst[e];
    float norm = dinv[s] * dinv[d];
    v4f hv = ((const v4f*)(H + (size_t)s * HID))[lane];     // coalesced b128 gather
    float* ad = Aout + (size_t)d * HID + lane * 4;
#pragma unroll
    for (int c = 0; c < 4; ++c) atomAddF(&ad[c], hv[c] * norm);
}

// ---------------- ReLU + f32 -> f16 for next WMMA GEMM (float4) ----------
__global__ void k_relu_half(const float* __restrict__ Aout, _Float16* __restrict__ Hh) {
    int t = blockIdx.x * blockDim.x + threadIdx.x;
    if (t >= NODES * (HID / 4)) return;
    v4f a = ((const v4f*)Aout)[t];
    v4h h;
#pragma unroll
    for (int i = 0; i < 4; ++i) h[i] = (_Float16)fmaxf(a[i], 0.f);
    ((v4h*)Hh)[t] = h;
}

// ---------------- pooling -------------------------------------------------
__global__ void k_pool_zero(float* sums, float* cnt) {
    int t = blockIdx.x * blockDim.x + threadIdx.x;
    if (t < NGRAPH * HID) sums[t] = 0.f;
    if (t < NGRAPH) cnt[t] = 0.f;
}
__global__ void k_pool_accum(const float* __restrict__ Aout, const int* __restrict__ batch,
                             float* __restrict__ sums, float* __restrict__ cnt) {
    int t = blockIdx.x * blockDim.x + threadIdx.x;
    if (t >= NODES * (HID / 4)) return;
    int node = t >> 5, j4 = t & 31;
    int g = batch[node];
    v4f a = ((const v4f*)Aout)[t];
    float* sg = sums + (size_t)g * HID + j4 * 4;
#pragma unroll
    for (int i = 0; i < 4; ++i) atomAddF(&sg[i], a[i]);
    if (j4 == 0) atomAddF(&cnt[g], 1.0f);
}
__global__ void k_final(const float* __restrict__ sums, const float* __restrict__ cnt,
                        const float* __restrict__ Wl, const float* __restrict__ bl,
                        float* __restrict__ out) {
    int t = blockIdx.x * blockDim.x + threadIdx.x;
    if (t >= NGRAPH * NCLS) return;
    int g = t / NCLS, c = t % NCLS;
    float inv = 1.f / fmaxf(cnt[g], 1.f);
    const float* sg = sums + (size_t)g * HID;
    float s = bl[c];
#pragma unroll 8
    for (int k = 0; k < HID; ++k) s = fmaf(sg[k] * inv, Wl[k * NCLS + c], s);
    out[t] = s;
}

extern "C" void kernel_launch(void* const* d_in, const int* in_sizes, int n_in,
                              void* d_out, int out_size, void* d_ws, size_t ws_size,
                              hipStream_t stream) {
    const float* x   = (const float*)d_in[0];
    const int*   ei  = (const int*)d_in[1];     // [2 x E]: row0=src, row1=dst
    const int*   bat = (const int*)d_in[2];
    const float* W1  = (const float*)d_in[3];
    const float* b1  = (const float*)d_in[4];
    const float* W2  = (const float*)d_in[5];
    const float* b2  = (const float*)d_in[6];
    const float* W3  = (const float*)d_in[7];
    const float* b3  = (const float*)d_in[8];
    const float* Wl  = (const float*)d_in[9];
    const float* bl  = (const float*)d_in[10];
    float* out = (float*)d_out;
    const int* src = ei;
    const int* dst = ei + EDGES;

    // ---- carve workspace ----
    char* ws = (char*)d_ws;
    size_t off = 0;
    auto carve = [&](size_t bytes) -> void* {
        off = (off + 255) & ~(size_t)255;
        void* p = ws + off;
        off += bytes;
        return p;
    };
    float*    deg   = (float*)carve(NODES * 4);
    float*    dinv  = (float*)carve(NODES * 4);
    float*    Hd    = (float*)carve((size_t)NODES * HID * 4);   // dense GEMM output
    float*    Aout  = (float*)carve((size_t)NODES * HID * 4);   // aggregated output
    _Float16* Hh    = (_Float16*)carve((size_t)NODES * HID * 2);// f16 activations
    _Float16* Bp    = (_Float16*)carve(HID * HID * 2);          // packed weights
    float*    sums  = (float*)carve((size_t)NGRAPH * HID * 4);
    float*    cnt   = (float*)carve(NGRAPH * 4);
    (void)ws_size; (void)n_in; (void)in_sizes; (void)out_size;

    const int B = 256;
    const int gNode   = (NODES + B - 1) / B;
    const int gEdge   = (EDGES + B - 1) / B;
    const int gNH     = (NODES * HID + B - 1) / B;
    const int gNH4    = (NODES * (HID / 4) + B - 1) / B;
    const int gEdge32 = (EDGES * 32 + B - 1) / B;
    const int mtiles  = NODES / 16;                       // 3125
    const int gGemm   = (mtiles + 7) / 8;                 // 8 waves/block

    // degrees (shared by all layers)
    k_deg_init<<<gNode, B, 0, stream>>>(deg);
    k_deg_count<<<gEdge, B, 0, stream>>>(dst, deg);
    k_dinv<<<gNode, B, 0, stream>>>(deg, dinv);

    // ---- layer 1 ----
    k_gemm_l1<<<gNH, B, 0, stream>>>(x, W1, Hd);
    k_self_bias<<<gNH4, B, 0, stream>>>(Hd, dinv, b1, Aout);
    k_edge_agg<<<gEdge32, B, 0, stream>>>(src, dst, dinv, Hd, Aout);
    k_relu_half<<<gNH4, B, 0, stream>>>(Aout, Hh);

    // ---- layer 2 ----
    k_pack_b<<<(HID * HID + B - 1) / B, B, 0, stream>>>(W2, Bp);
    k_gemm_wmma<<<gGemm, B, 0, stream>>>(Hh, Bp, Hd, mtiles);
    k_self_bias<<<gNH4, B, 0, stream>>>(Hd, dinv, b2, Aout);
    k_edge_agg<<<gEdge32, B, 0, stream>>>(src, dst, dinv, Hd, Aout);
    k_relu_half<<<gNH4, B, 0, stream>>>(Aout, Hh);

    // ---- layer 3 (no ReLU) ----
    k_pack_b<<<(HID * HID + B - 1) / B, B, 0, stream>>>(W3, Bp);
    k_gemm_wmma<<<gGemm, B, 0, stream>>>(Hh, Bp, Hd, mtiles);
    k_self_bias<<<gNH4, B, 0, stream>>>(Hd, dinv, b3, Aout);
    k_edge_agg<<<gEdge32, B, 0, stream>>>(src, dst, dinv, Hd, Aout);

    // ---- mean pool + linear head ----
    k_pool_zero<<<(NGRAPH * HID + B - 1) / B, B, 0, stream>>>(sums, cnt);
    k_pool_accum<<<gNH4, B, 0, stream>>>(Aout, bat, sums, cnt);
    k_final<<<(NGRAPH * NCLS + B - 1) / B, B, 0, stream>>>(sums, cnt, Wl, bl, out);
}
